// BNNLeNetMnist_28518582846028
// MI455X (gfx1250) — 8-bit matrix operand order repaired and checked by reading, compiled, not run on hardware
//
#include <hip/hip_runtime.h>
#include <math.h>

typedef int v8i __attribute__((ext_vector_type(8)));

#define BNN_EPS 1e-5f
#define BATCH 16384

__device__ __forceinline__ int imax2(int a, int b) { return a > b ? a : b; }
__device__ __forceinline__ int imin2(int a, int b) { return a < b ? a : b; }

__global__ void conv1_pool_bn_sign(const float* __restrict__ x,
                                   const float* __restrict__ w1,
                                   const float* __restrict__ b1,
                                   const float* __restrict__ g,
                                   const float* __restrict__ bb,
                                   const float* __restrict__ mm,
                                   const float* __restrict__ vv,
                                   signed char* __restrict__ act1,
                                   int total) {
  int tid = blockIdx.x * blockDim.x + threadIdx.x;
  if (tid >= total) return;
  int px = tid % 14;
  int py = (tid / 14) % 14;
  int c  = (tid / 196) % 6;
  int b  = tid / 1176;

  float ws[25];
#pragma unroll
  for (int j = 0; j < 25; ++j) ws[j] = (w1[c * 25 + j] >= 0.f) ? 1.f : -1.f;

  const float* xb = x + (size_t)b * 784;
  float win[6][6];
#pragma unroll
  for (int r = 0; r < 6; ++r) {
    int iy = 2 * py + r - 2;
    bool ry = (iy >= 0) && (iy < 28);
    int iyc = imin2(imax2(iy, 0), 27);
#pragma unroll
    for (int cx = 0; cx < 6; ++cx) {
      int ix = 2 * px + cx - 2;
      bool rx = (ix >= 0) && (ix < 28);
      int ixc = imin2(imax2(ix, 0), 27);
      float v = xb[iyc * 28 + ixc];
      win[r][cx] = (ry && rx) ? v : 0.f;
    }
  }

  float best = -3.4e38f;
#pragma unroll
  for (int q = 0; q < 4; ++q) {
    float s = 0.f;
#pragma unroll
    for (int dy = 0; dy < 5; ++dy)
#pragma unroll
      for (int dx = 0; dx < 5; ++dx)
        s += ws[dy * 5 + dx] * win[dy + (q >> 1)][dx + (q & 1)];
    best = fmaxf(best, s);
  }
  float inv = g[c] * rsqrtf(vv[c] + BNN_EPS);
  float u = (best + b1[c]) * inv + (bb[c] - mm[c] * inv);
  u = fminf(fmaxf(u, -1.f), 1.f);
  act1[tid] = (u >= 0.f) ? (signed char)1 : (signed char)-1;
}

__global__ void pack_wsign(const float* __restrict__ w,
                           signed char* __restrict__ out,
                           int N, int K, int kchunks, int total) {
  int tid = blockIdx.x * blockDim.x + threadIdx.x;
  if (tid >= total) return;
  int byteIdx = tid & 1023;
  int f    = tid >> 10;
  int lane = byteIdx >> 5;
  int byte = byteIdx & 31;
  int v  = byte >> 2;
  int bb = byte & 3;
  int nt = f / kchunks;
  int kc = f % kchunks;
  int n = nt * 16 + (lane & 15);
  int k = kc * 64 + (v >> 1) * 16 + (lane >> 4) * 8 + (v & 1) * 4 + bb;
  signed char val = 0;
  if (n < N && k < K) val = (w[(size_t)n * K + k] >= 0.f) ? 1 : -1;
  out[tid] = val;
}

__global__ void gemm_bin_conv2(const signed char* __restrict__ act1,
                               const signed char* __restrict__ Bp,
                               const float* __restrict__ b2,
                               const float* __restrict__ g,
                               const float* __restrict__ bb_,
                               const float* __restrict__ mm,
                               const float* __restrict__ vv,
                               signed char* __restrict__ act2,
                               int tiles) {
  __shared__ int ldsb[3 * 256];
  __shared__ signed char ldsa[8][16 * 192];

  for (int i = threadIdx.x; i < 3 * 256; i += blockDim.x)
    ldsb[i] = ((const int*)Bp)[i];

  int wid  = threadIdx.x >> 5;
  int lane = threadIdx.x & 31;
  int raw  = blockIdx.x * (blockDim.x >> 5) + wid;
  bool valid = raw < tiles;
  int tile = imin2(raw, tiles - 1);

  int m = lane & 15;
  int Mrow = tile * 16 + m;
  int P = Mrow >> 2;
  int q = Mrow & 3;
  int b   = P / 49;
  int r49 = P % 49;
  int py = r49 / 7, px = r49 % 7;
  int y  = 2 * py + (q >> 1);
  int xx = 2 * px + (q & 1);
  const signed char* basep = act1 + (size_t)b * 1176;

  signed char* atile = &ldsa[wid][0];
  int kbase = (lane >> 4) * 96;
  for (int i = 0; i < 24; ++i) {
    int k0 = kbase + i * 4;
    int word = 0;
#pragma unroll
    for (int t = 0; t < 4; ++t) {
      int k = k0 + t;
      int c  = imin2(k / 25, 5);
      int rr = k - c * 25;
      int dy = rr / 5;
      int dx = rr - dy * 5;
      int iy = y + dy - 2, ix = xx + dx - 2;
      int iyc = imin2(imax2(iy, 0), 13);
      int ixc = imin2(imax2(ix, 0), 13);
      int s = (int)basep[c * 196 + iyc * 14 + ixc];
      bool ok = (k < 150) && (iy == iyc) && (ix == ixc);
      word |= (ok ? (s & 0xff) : 0) << (8 * t);
    }
    *(int*)(atile + m * 192 + k0) = word;
  }
  __syncthreads();

  int koff_half = (lane >> 4) << 3;
  v8i acc = {0, 0, 0, 0, 0, 0, 0, 0};
#pragma unroll
  for (int kc = 0; kc < 3; ++kc) {
    v8i afrag;
#pragma unroll
    for (int p = 0; p < 4; ++p) {
      int k = kc * 64 + p * 16 + koff_half;
      int2 d = *(const int2*)(atile + m * 192 + k);
      afrag[2 * p]     = d.x;
      afrag[2 * p + 1] = d.y;
    }
    const int* bp = &ldsb[kc * 256 + lane * 8];
    int4 q0 = *(const int4*)bp;
    int4 q1 = *(const int4*)(bp + 4);
    v8i bfrag;
    bfrag[0] = q0.x; bfrag[1] = q0.y; bfrag[2] = q0.z; bfrag[3] = q0.w;
    bfrag[4] = q1.x; bfrag[5] = q1.y; bfrag[6] = q1.z; bfrag[7] = q1.w;

    acc = __builtin_amdgcn_wmma_i32_16x16x64_iu8(true, afrag, true, bfrag, acc,
                                                 false, false);
  }

  if (valid) {
    int n = lane & 15;
    float inv = g[n] * rsqrtf(vv[n] + BNN_EPS);
    float sh  = bb_[n] - mm[n] * inv;
    float bs  = b2[n];
    int g0 = tile * 4 + ((lane >> 4) << 1);
#pragma unroll
    for (int h = 0; h < 2; ++h) {
      int gi = g0 + h;
      int vmax = acc[h * 4 + 0];
      vmax = imax2(vmax, acc[h * 4 + 1]);
      vmax = imax2(vmax, acc[h * 4 + 2]);
      vmax = imax2(vmax, acc[h * 4 + 3]);
      int ob  = gi / 49;
      int orr = gi % 49;
      int opy = orr / 7, opx = orr % 7;
      float u = ((float)vmax + bs) * inv + sh;
      u = fminf(fmaxf(u, -1.f), 1.f);
      act2[(size_t)ob * 832 + n * 49 + opy * 7 + opx] =
          (u >= 0.f) ? (signed char)1 : (signed char)-1;
    }
  }
}

__global__ void gemm_bin_fc(const signed char* __restrict__ A, int Kp,
                            int kchunks, const signed char* __restrict__ Bp,
                            int tiles_m,
                            const float* __restrict__ bias,
                            const float* __restrict__ g,
                            const float* __restrict__ bb_,
                            const float* __restrict__ mm,
                            const float* __restrict__ vv,
                            int n_valid, int mode,
                            signed char* __restrict__ out_i8, int out_stride,
                            int* __restrict__ out_i32) {
  __shared__ int ldsb[13 * 256];
  int tile_n = blockIdx.y;
  {
    const int* src = (const int*)(Bp + (size_t)tile_n * kchunks * 1024);
    int nInts = kchunks * 256;
    for (int i = threadIdx.x; i < nInts; i += blockDim.x) ldsb[i] = src[i];
  }
  __syncthreads();

  int wid  = threadIdx.x >> 5;
  int lane = threadIdx.x & 31;
  int tile_m = blockIdx.x * (blockDim.x >> 5) + wid;
  if (tile_m >= tiles_m) return;

  int m = tile_m * 16 + (lane & 15);
  const signed char* arow = A + (size_t)m * Kp;
  int koff_half = (lane >> 4) << 3;

  v8i acc = {0, 0, 0, 0, 0, 0, 0, 0};
  for (int kc = 0; kc < kchunks; ++kc) {
    __builtin_prefetch(arow + (kc + 1) * 64, 0, 3);
    v8i afrag;
#pragma unroll
    for (int p = 0; p < 4; ++p) {
      int k = kc * 64 + p * 16 + koff_half;
      int2 d = *(const int2*)(arow + k);
      afrag[2 * p]     = d.x;
      afrag[2 * p + 1] = d.y;
    }
    const int* bp = &ldsb[kc * 256 + lane * 8];
    int4 q0 = *(const int4*)bp;
    int4 q1 = *(const int4*)(bp + 4);
    v8i bfrag;
    bfrag[0] = q0.x; bfrag[1] = q0.y; bfrag[2] = q0.z; bfrag[3] = q0.w;
    bfrag[4] = q1.x; bfrag[5] = q1.y; bfrag[6] = q1.z; bfrag[7] = q1.w;

    acc = __builtin_amdgcn_wmma_i32_16x16x64_iu8(true, afrag, true, bfrag, acc,
                                                 false, false);
  }

  int nl = lane & 15;
  int n  = tile_n * 16 + nl;
  int rbase = (lane >> 4) << 3;
  if (mode == 0) {
    if (n < n_valid) {
      float inv = g[n] * rsqrtf(vv[n] + BNN_EPS);
      float sh  = bb_[n] - mm[n] * inv;
      float bs  = bias[n];
#pragma unroll
      for (int r = 0; r < 8; ++r) {
        int mrow = tile_m * 16 + r + rbase;
        float u = ((float)acc[r] + bs) * inv + sh;
        u = fminf(fmaxf(u, -1.f), 1.f);
        out_i8[(size_t)mrow * out_stride + n] =
            (u >= 0.f) ? (signed char)1 : (signed char)-1;
      }
    }
  } else {
#pragma unroll
    for (int r = 0; r < 8; ++r) {
      int mrow = tile_m * 16 + r + rbase;
      out_i32[(size_t)mrow * 16 + nl] = acc[r];
    }
  }
}

__global__ void final_bn_logsoftmax(const int* __restrict__ fc3,
                                    const float* __restrict__ b5,
                                    const float* __restrict__ g,
                                    const float* __restrict__ bb_,
                                    const float* __restrict__ mm,
                                    const float* __restrict__ vv,
                                    float* __restrict__ out, int B) {
  int b = blockIdx.x * blockDim.x + threadIdx.x;
  if (b >= B) return;
  float z[10];
  float mx = -3.4e38f;
#pragma unroll
  for (int j = 0; j < 10; ++j) {
    float inv = g[j] * rsqrtf(vv[j] + BNN_EPS);
    float u = ((float)fc3[(size_t)b * 16 + j] + b5[j]) * inv +
              (bb_[j] - mm[j] * inv);
    z[j] = u;
    mx = fmaxf(mx, u);
  }
  float se = 0.f;
#pragma unroll
  for (int j = 0; j < 10; ++j) se += expf(z[j] - mx);
  float lse = mx + logf(se);
#pragma unroll
  for (int j = 0; j < 10; ++j) out[(size_t)b * 10 + j] = z[j] - lse;
}

extern "C" void kernel_launch(void* const* d_in, const int* in_sizes, int n_in,
                              void* d_out, int out_size, void* d_ws,
                              size_t ws_size, hipStream_t stream) {
  (void)in_sizes; (void)n_in; (void)out_size; (void)ws_size;
  const float* x  = (const float*)d_in[0];
  const float* w1 = (const float*)d_in[1];
  const float* b1 = (const float*)d_in[2];
  const float* w2 = (const float*)d_in[3];
  const float* b2 = (const float*)d_in[4];
  const float* w3 = (const float*)d_in[5];
  const float* b3 = (const float*)d_in[6];
  const float* w4 = (const float*)d_in[7];
  const float* b4 = (const float*)d_in[8];
  const float* w5 = (const float*)d_in[9];
  const float* b5 = (const float*)d_in[10];
  const float* bn1g = (const float*)d_in[11];
  const float* bn1b = (const float*)d_in[12];
  const float* bn1m = (const float*)d_in[13];
  const float* bn1v = (const float*)d_in[14];
  const float* bn2g = (const float*)d_in[15];
  const float* bn2b = (const float*)d_in[16];
  const float* bn2m = (const float*)d_in[17];
  const float* bn2v = (const float*)d_in[18];
  const float* bn3g = (const float*)d_in[19];
  const float* bn3b = (const float*)d_in[20];
  const float* bn3m = (const float*)d_in[21];
  const float* bn3v = (const float*)d_in[22];
  const float* bn4g = (const float*)d_in[23];
  const float* bn4b = (const float*)d_in[24];
  const float* bn4m = (const float*)d_in[25];
  const float* bn4v = (const float*)d_in[26];
  const float* bn5g = (const float*)d_in[27];
  const float* bn5b = (const float*)d_in[28];
  const float* bn5m = (const float*)d_in[29];
  const float* bn5v = (const float*)d_in[30];
  float* out = (float*)d_out;

  const int B = BATCH;
  char* ws = (char*)d_ws;
  size_t off = 0;
  auto carve = [&](size_t n) -> char* {
    off = (off + 255) & ~(size_t)255;
    char* p = ws + off;
    off += n;
    return p;
  };
  signed char* act1 = (signed char*)carve((size_t)B * 1176);
  signed char* act2 = (signed char*)carve((size_t)B * 832);
  signed char* act3 = (signed char*)carve((size_t)B * 128);
  signed char* act4 = (signed char*)carve((size_t)B * 128);
  int*         fcf  = (int*)carve((size_t)B * 16 * 4);
  signed char* pw2  = (signed char*)carve((size_t)1 * 3 * 1024);
  signed char* pw3  = (signed char*)carve((size_t)8 * 13 * 1024);
  signed char* pw4  = (signed char*)carve((size_t)6 * 2 * 1024);
  signed char* pw5  = (signed char*)carve((size_t)1 * 2 * 1024);
  size_t used = (off + 255) & ~(size_t)255;

  hipMemsetAsync(d_ws, 0, used, stream);

  {
    int total = B * 1176;
    conv1_pool_bn_sign<<<(total + 255) / 256, 256, 0, stream>>>(
        x, w1, b1, bn1g, bn1b, bn1m, bn1v, act1, total);
  }
  {
    int t2 = 1 * 3 * 1024;
    pack_wsign<<<(t2 + 255) / 256, 256, 0, stream>>>(w2, pw2, 16, 150, 3, t2);
    int t3 = 8 * 13 * 1024;
    pack_wsign<<<(t3 + 255) / 256, 256, 0, stream>>>(w3, pw3, 120, 784, 13, t3);
    int t4 = 6 * 2 * 1024;
    pack_wsign<<<(t4 + 255) / 256, 256, 0, stream>>>(w4, pw4, 84, 120, 2, t4);
    int t5 = 1 * 2 * 1024;
    pack_wsign<<<(t5 + 255) / 256, 256, 0, stream>>>(w5, pw5, 10, 84, 2, t5);
  }
  {
    int tiles = B * 196 / 16;
    gemm_bin_conv2<<<(tiles + 7) / 8, 256, 0, stream>>>(
        act1, pw2, b2, bn2g, bn2b, bn2m, bn2v, act2, tiles);
  }
  {
    int tiles_m = B / 16;
    dim3 grid((tiles_m + 7) / 8, 8);
    gemm_bin_fc<<<grid, 256, 0, stream>>>(act2, 832, 13, pw3, tiles_m, b3,
                                          bn3g, bn3b, bn3m, bn3v, 120, 0,
                                          act3, 128, nullptr);
  }
  {
    int tiles_m = B / 16;
    dim3 grid((tiles_m + 7) / 8, 6);
    gemm_bin_fc<<<grid, 256, 0, stream>>>(act3, 128, 2, pw4, tiles_m, b4,
                                          bn4g, bn4b, bn4m, bn4v, 84, 0,
                                          act4, 128, nullptr);
  }
  {
    int tiles_m = B / 16;
    dim3 grid((tiles_m + 7) / 8, 1);
    gemm_bin_fc<<<grid, 256, 0, stream>>>(act4, 128, 2, pw5, tiles_m, b5,
                                          bn5g, bn5b, bn5m, bn5v, 10, 1,
                                          nullptr, 0, fcf);
  }
  final_bn_logsoftmax<<<(B + 255) / 256, 256, 0, stream>>>(
      fcf, b5, bn5g, bn5b, bn5m, bn5v, out, B);
}
